// ScaledDotProductAttention_62277025792399
// MI455X (gfx1250) — matrix operand order repaired and checked by reading, not compiled, not run on hardware
//
#include <hip/hip_runtime.h>

// CDNA5 WMMA operand types + a true vector float4 (HIP's struct float4 can't
// be used with __builtin_nontemporal_store)
typedef __attribute__((ext_vector_type(16))) _Float16 v16h;
typedef __attribute__((ext_vector_type(8)))  float    v8f;
typedef __attribute__((ext_vector_type(4)))  float    v4f;

#define SEQ_L    2048
#define D_HEAD   64
#define Q_BLK    16
#define SROW     2052          // 2048 + 4 pad -> conflict-free column access (2052 % 64 == 4)
#define NWAVES   8
#define NTHREADS 256
#define NHEADS   32            // B*N = 2*16

__global__ __launch_bounds__(NTHREADS)
void sdpa_kernel(const float* __restrict__ Q,
                 const float* __restrict__ K,
                 const float* __restrict__ V,
                 const unsigned char* __restrict__ M,   // jnp.bool_ = 1 byte
                 float* __restrict__ Out,
                 float* __restrict__ Attn)
{
    extern __shared__ float lds[];
    float* sS = lds;                        // [Q_BLK][SROW]  masked scores -> exp(s-max)
    float* sO = lds + Q_BLK * SROW;         // [2][Q_BLK][D_HEAD] PV partials
    float* sR = sO + 2 * Q_BLK * D_HEAD;    // [Q_BLK] 1/rowsum

    const int tid  = threadIdx.x;
    const int lane = tid & 31;
    const int wave = tid >> 5;
    const int ln   = lane & 15;             // row/col within 16
    const int hi   = lane >> 4;             // half-wave select

    const int gid = blockIdx.x;
    const int bh  = gid >> 7;               // head index, 128 q-blocks per head
    const int q0  = (gid & 127) * Q_BLK;

    const size_t headQKV = (size_t)bh * SEQ_L * D_HEAD;
    const float* Qb = Q + headQKV + (size_t)q0 * D_HEAD;
    const float* Kb = K + headQKV;
    const float* Vb = V + headQKV;
    const unsigned char* Mb = M + (size_t)bh * SEQ_L * SEQ_L + (size_t)q0 * SEQ_L;
    float* Ab = Attn + (size_t)bh * SEQ_L * SEQ_L + (size_t)q0 * SEQ_L;
    float* Ob = Out  + headQKV + (size_t)q0 * D_HEAD;

    // ---- Q tile -> WMMA A-operand registers (f32 -> f16), reused by all K-tiles ----
    // A 16x32 k order packed here (contiguous, not AMD's table order): lanes 0-15 row m, K 0..15; lanes 16-31: K 16..31 (matches B)
    v16h aQ[2];
    {
        const float* qrow = Qb + (size_t)ln * D_HEAD;
#pragma unroll
        for (int c = 0; c < 2; ++c) {
            const float* p = qrow + c * 32 + hi * 16;
            const v4f q0v = *(const v4f*)(p);
            const v4f q1v = *(const v4f*)(p + 4);
            const v4f q2v = *(const v4f*)(p + 8);
            const v4f q3v = *(const v4f*)(p + 12);
            v16h a;
#pragma unroll
            for (int j = 0; j < 4; ++j) {
                a[j]      = (_Float16)q0v[j];
                a[j + 4]  = (_Float16)q1v[j];
                a[j + 8]  = (_Float16)q2v[j];
                a[j + 12] = (_Float16)q3v[j];
            }
            aQ[c] = a;
        }
    }

    // ---------------- Phase 1: S = mask(Q K^T / 8) into LDS ----------------
    for (int t = wave; t < SEQ_L / 16; t += NWAVES) {
        const int k0 = t * 16;
        v8f acc = {};
#pragma unroll
        for (int c = 0; c < 2; ++c) {
            // B 32x16 as packed here (same k order as A): lane holds column n = ln; lanes 0-15 K 0..15, 16-31 K 16..31.
            // Column n of B == row (k0+n) of K-matrix -> contiguous b128 loads.
            const float* kr = Kb + (size_t)(k0 + ln) * D_HEAD + c * 32 + hi * 16;
            v16h b;
#pragma unroll
            for (int j = 0; j < 4; ++j) {
                const v4f f = *(const v4f*)(kr + 4 * j);
                b[4*j+0] = (_Float16)f[0]; b[4*j+1] = (_Float16)f[1];
                b[4*j+2] = (_Float16)f[2]; b[4*j+3] = (_Float16)f[3];
            }
            acc = __builtin_amdgcn_wmma_f32_16x16x32_f16(false, aQ[c], false, b,
                                                         (short)0, acc, false, false);
        }
        // C/D layout: VGPR r -> row r (lanes 0-15) / row r+8 (lanes 16-31), col = ln
#pragma unroll
        for (int r = 0; r < 8; ++r) {
            const int row = r + hi * 8;
            float s = acc[r] * 0.125f;                       // 1/TEMPERATURE
            if (Mb[(size_t)row * SEQ_L + k0 + ln]) s = -__builtin_inff();
            sS[row * SROW + k0 + ln] = s;
        }
    }
    __syncthreads();

    // ---------------- Phase 2: row softmax (unnormalized) in LDS ----------------
    {
        const int row = tid >> 4;
        const int j   = tid & 15;
        float* rowp = sS + row * SROW;
        float mx = -__builtin_inff();
        for (int c = j; c < SEQ_L; c += 16) mx = fmaxf(mx, rowp[c]);
#pragma unroll
        for (int m = 1; m < 16; m <<= 1) mx = fmaxf(mx, __shfl_xor(mx, m, 32));
        float sum = 0.f;
        for (int c = j; c < SEQ_L; c += 16) {
            const float s = rowp[c];
            const float e = (s == -__builtin_inff()) ? 0.f : __expf(s - mx);
            rowp[c] = e;                 // keep UNNORMALIZED e; 1/sum folded in later
            sum += e;
        }
#pragma unroll
        for (int m = 1; m < 16; m <<= 1) sum += __shfl_xor(sum, m, 32);
        if (j == 0) sR[row] = (sum > 0.f) ? 1.f / sum : 0.f;  // all-masked row -> 0
    }
    __syncthreads();

    // ---- stream attn = e * (1/sum) to HBM: b128 non-temporal stores, LDS read-only ----
#pragma unroll 1
    for (int row = 0; row < Q_BLK; ++row) {
        const float rinv = sR[row];
        const v4f* rowp = (const v4f*)(sS + row * SROW);     // row base is 16B aligned
        v4f* gout = (v4f*)(Ab + (size_t)row * SEQ_L);
        for (int c = tid; c < SEQ_L / 4; c += NTHREADS)
            __builtin_nontemporal_store(rowp[c] * rinv, gout + c);
    }
    // no barrier needed: phase 3 also only READS sS/sR

    // ---------------- Phase 3: O = (1/sum) * (e @ V), P from LDS, V from L2 ----------------
    {
        const int ntile = wave & 3;      // which 16 columns of D
        const int khalf = wave >> 2;     // which half of the 2048-long reduction
        v8f oacc = {};
        for (int ch = 0; ch < 32; ++ch) {
            const int kc = khalf * 1024 + ch * 32;
            // A = e chunk 16x32 from LDS via b128 (padded stride -> <=2-way conflicts)
            const float* pr = sS + ln * SROW + kc + hi * 16;
            const v4f p0 = *(const v4f*)(pr);
            const v4f p1 = *(const v4f*)(pr + 4);
            const v4f p2 = *(const v4f*)(pr + 8);
            const v4f p3 = *(const v4f*)(pr + 12);
            v16h a;
#pragma unroll
            for (int j = 0; j < 4; ++j) {
                a[j]      = (_Float16)p0[j];
                a[j + 4]  = (_Float16)p1[j];
                a[j + 8]  = (_Float16)p2[j];
                a[j + 12] = (_Float16)p3[j];
            }
            // B = V chunk 32x16: lane = column d, K contiguous per half-wave
            const float* vp = Vb + (size_t)(kc + hi * 16) * D_HEAD + ntile * 16 + ln;
            v16h b;
#pragma unroll
            for (int kk = 0; kk < 16; ++kk) b[kk] = (_Float16)vp[(size_t)kk * D_HEAD];
            oacc = __builtin_amdgcn_wmma_f32_16x16x32_f16(false, a, false, b,
                                                          (short)0, oacc, false, false);
        }
        // fold per-row 1/sum into the accumulator rows, then store exclusive partials
        float* op = sO + khalf * Q_BLK * D_HEAD;
#pragma unroll
        for (int r = 0; r < 8; ++r) {
            const int row = r + hi * 8;
            op[row * D_HEAD + ntile * 16 + ln] = oacc[r] * sR[row];
        }
    }
    __syncthreads();

    // 2-way reduce of PV partials + b128 streaming write of output
    {
        const v4f* p0 = (const v4f*)sO;
        const v4f* p1 = (const v4f*)(sO + Q_BLK * D_HEAD);
        v4f* gout = (v4f*)Ob;
        const int i = tid;                                   // 256 float4 = 16x64 floats
        __builtin_nontemporal_store(p0[i] + p1[i], gout + i);
    }
}

extern "C" void kernel_launch(void* const* d_in, const int* in_sizes, int n_in,
                              void* d_out, int out_size, void* d_ws, size_t ws_size,
                              hipStream_t stream) {
    const float*         q = (const float*)d_in[0];
    const float*         k = (const float*)d_in[1];
    const float*         v = (const float*)d_in[2];
    const unsigned char* m = (const unsigned char*)d_in[3];   // bool mask, 1 byte/elem

    float* out  = (float*)d_out;
    float* attn = out + (size_t)NHEADS * SEQ_L * D_HEAD;      // tuple order: (output, attn)

    const size_t shmem = (Q_BLK * SROW + 2 * Q_BLK * D_HEAD + Q_BLK) * sizeof(float);
    hipFuncSetAttribute(reinterpret_cast<const void*>(sdpa_kernel),
                        hipFuncAttributeMaxDynamicSharedMemorySize, (int)shmem);

    const int blocks = NHEADS * (SEQ_L / Q_BLK);              // 4096 workgroups
    sdpa_kernel<<<blocks, NTHREADS, shmem, stream>>>(q, k, v, m, out, attn);
}
